// SchNetAuTopology_42769284334265
// MI455X (gfx1250) — matrix operand order repaired and checked by reading, not compiled, not run on hardware
//
#include <hip/hip_runtime.h>
#include <stdint.h>

#define N_ATOMS 50000
#define N_EDGES 600000
#define N_FEAT 128
#define N_GAUSS 32
#define ATOMS_PER_MOL 50
#define N_MOL (N_ATOMS / ATOMS_PER_MOL)
#define CUTOFF 5.0f
#define LOG2F_C 0.69314718056f

// padded K-strides for transposed LDS weight tiles (bank-conflict avoidance)
#define LDK128 136  // 128 + 8
#define LDK32  40   // 32 + 8

typedef __attribute__((ext_vector_type(16))) __bf16 v16bf;
typedef __attribute__((ext_vector_type(8)))  __bf16 v8bf;
typedef __attribute__((ext_vector_type(8)))  float  v8f;

typedef __attribute__((ext_vector_type(4))) unsigned int u32x4;
typedef __attribute__((ext_vector_type(4))) int          i32x4;
typedef __attribute__((ext_vector_type(8))) int          i32x8;

#if defined(__HIP_DEVICE_COMPILE__) && __has_builtin(__builtin_amdgcn_tensor_load_to_lds) && \
    __has_builtin(__builtin_amdgcn_s_wait_tensorcnt)
#define HAS_TDM 1
#else
#define HAS_TDM 0
#endif

static __device__ __forceinline__ float ssp_f(float x) {
  return fmaxf(x, 0.0f) + log1pf(__expf(-fabsf(x))) - LOG2F_C;
}

static __device__ __forceinline__ v8f wmma_bf16(v16bf a, v16bf b, v8f c) {
  return __builtin_amdgcn_wmma_f32_16x16x32_bf16(false, a, false, b, (short)0, c,
                                                 false, false);
}

static __device__ __forceinline__ v8f v8f_zero() {
  v8f z = {0.f, 0.f, 0.f, 0.f, 0.f, 0.f, 0.f, 0.f};
  return z;
}

static __device__ __forceinline__ v16bf cat8(v8bf lo, v8bf hi) {
  return __builtin_shufflevector(lo, hi, 0, 1, 2, 3, 4, 5, 6, 7, 8, 9, 10, 11,
                                 12, 13, 14, 15);
}

// A fragment (16x32) from row-major bf16: two contiguous 8-elem K-runs -> b128 x2
static __device__ __forceinline__ v16bf load_A_bf16(const __bf16* p, int ld,
                                                    int lane) {
  const int half = (lane >> 4) & 1;
  const int row  = lane & 15;
  const __bf16* rp = p + row * ld + half * 8;
  return cat8(*(const v8bf*)rp, *(const v8bf*)(rp + 16));
}

// A fragment from row-major f32 (contiguous 8-float runs -> b128 + cvt_pk)
static __device__ __forceinline__ v16bf load_A_f32(const float* p, int ld,
                                                   int lane) {
  const int half = (lane >> 4) & 1;
  const int row  = lane & 15;
  const float* rp = p + row * ld + half * 8;
  v16bf a;
#pragma unroll
  for (int j = 0; j < 8; ++j) {
    a[j]     = (__bf16)rp[j];
    a[8 + j] = (__bf16)rp[16 + j];
  }
  return a;
}

// B fragment (32x16) from *transposed* (N-major) LDS tile with K-stride ldK:
// lane's 16 values are two 8-elem K-runs (k0+8h, k0+16+8h) -> two aligned ds_load_b128.
static __device__ __forceinline__ v16bf load_B_T(const __bf16* sWt, int ldK,
                                                 int ncol0, int k0, int lane) {
  const int khalf = (lane >> 4) & 1;
  const int col   = lane & 15;
  const __bf16* src = sWt + (ncol0 + col) * ldK + k0 + khalf * 8;
  return cat8(*(const v8bf*)src, *(const v8bf*)(src + 16));
}

// ---- weight tile staging: TDM DMA (with HW pad) or manual fallback ----

// dense transposed weight (rows x K bf16) -> padded LDS tile, manual path
static __device__ __forceinline__ void lds_copy_w(__bf16* dst, const __bf16* src,
                                                  int rows, int rowsU4, int ldU4) {
  uint4* d = (uint4*)dst;
  const uint4* s = (const uint4*)src;
  const int total = rows * rowsU4;
  for (int i = threadIdx.x; i < total; i += 128) {
    int n = i / rowsU4, j = i - n * rowsU4;
    d[n * ldU4 + j] = s[i];
  }
}

#if HAS_TDM
// Issue one TDM transfer: rows x K bf16 tile, LDS padded by 4 DWORDs (8 bf16)
// every K elements.  pad_code: K=128 -> 5 (64 DW), K=32 -> 3 (16 DW).
// Call from a single wave; follow with s_wait_tensorcnt + block barrier.
static __device__ __forceinline__ void tdm_load_tile(const void* gsrc,
                                                     void* lds_dst, int K,
                                                     int rows, int pad_code) {
  const uint64_t ga = (uint64_t)(uintptr_t)gsrc;
  const uint32_t la = (uint32_t)(uintptr_t)lds_dst;  // low 32b = LDS byte offset
  u32x4 g0;
  g0[0] = 1u;                                          // count=1, user mode
  g0[1] = la;                                          // lds_addr
  g0[2] = (uint32_t)ga;                                // global_addr[31:0]
  g0[3] = (uint32_t)((ga >> 32) & 0x1FFFFFFu) | (2u << 30);  // addr hi | type=2
  i32x8 g1;
  g1[0] = (int)((1u << 16) |                 // data_size = 2 bytes
                (1u << 20) |                 // pad_enable
                ((uint32_t)pad_code << 22) | // pad_interval
                (3u << 25));                 // pad_amount = 4 DWORDs
  g1[1] = (int)((uint32_t)K << 16);          // tensor_dim0[15:0] @ [31:16]
  g1[2] = (int)(((uint32_t)rows) << 16);     // tensor_dim1[15:0] @ [31:16]
  g1[3] = (int)((uint32_t)K << 16);          // tile_dim0 @ [31:16]
  g1[4] = (int)((uint32_t)rows);             // tile_dim1 ; tile_dim2=0
  g1[5] = (int)K;                            // tensor_dim0_stride[31:0]
  g1[6] = 0;
  g1[7] = 0;
  i32x4 z4 = {0, 0, 0, 0};
#if __clang_major__ >= 23
  i32x8 z8 = {0, 0, 0, 0, 0, 0, 0, 0};
  __builtin_amdgcn_tensor_load_to_lds(g0, g1, z4, z4, z8, 0);
#else
  __builtin_amdgcn_tensor_load_to_lds(g0, g1, z4, z4, 0);
#endif
}
#define STAGE_W(dst, src, rows, rowsU4, ldU4, K, padc)            \
  do {                                                            \
    if (threadIdx.x < 32) {                                       \
      tdm_load_tile((src), (dst), (K), (rows), (padc));           \
      __builtin_amdgcn_s_wait_tensorcnt(0);                       \
    }                                                             \
  } while (0)
#else
#define STAGE_W(dst, src, rows, rowsU4, ldU4, K, padc) \
  lds_copy_w((dst), (src), (rows), (rowsU4), (ldU4))
#endif

// ---------------- small utility kernels ----------------

__global__ void k_zero(float* __restrict__ p, int n) {
  int i = blockIdx.x * 256 + threadIdx.x;
  if (i < n) p[i] = 0.0f;
}

// f32 (L,K,N) -> bf16 transposed (L,N,K), dense
__global__ void k_cvt_t(const float* __restrict__ s, __bf16* __restrict__ d,
                        int K, int N, int total) {
  int i = blockIdx.x * 256 + threadIdx.x;
  if (i >= total) return;
  int kn = K * N;
  int l = i / kn;
  int rem = i - l * kn;
  int k = rem / N, n = rem - k * N;
  d[l * kn + n * K + k] = (__bf16)s[i];
}

__global__ void k_init_r(const int* __restrict__ z, const float* __restrict__ embed,
                         float* __restrict__ r) {
  int i = blockIdx.x * 256 + threadIdx.x;
  if (i < N_ATOMS * N_FEAT) {
    int atom = i >> 7, f = i & 127;
    r[i] = embed[z[atom] * N_FEAT + f];
  }
}

__global__ void k_gauss(const float* __restrict__ xyz, const int* __restrict__ nbr,
                        __bf16* __restrict__ gauss) {
  int e = blockIdx.x * 256 + threadIdx.x;
  if (e >= N_EDGES) return;
  int a0 = nbr[2 * e], a1 = nbr[2 * e + 1];
  float dx = xyz[3 * a0 + 0] - xyz[3 * a1 + 0];
  float dy = xyz[3 * a0 + 1] - xyz[3 * a1 + 1];
  float dz = xyz[3 * a0 + 2] - xyz[3 * a1 + 2];
  float d = sqrtf(dx * dx + dy * dy + dz * dz);
  const float width = CUTOFF / (float)(N_GAUSS - 1);
  const float coef  = -0.5f / (width * width);
  __bf16* gp = gauss + (size_t)e * N_GAUSS;
#pragma unroll
  for (int g = 0; g < N_GAUSS; ++g) {
    float t = d - width * (float)g;
    gp[g] = (__bf16)__expf(coef * t * t);
  }
}

// ---------------- rf = r @ Wn + bn  (stored bf16) ----------------

__global__ void __launch_bounds__(128) k_rf(const float* __restrict__ r,
                                            const __bf16* __restrict__ Wt,
                                            const float* __restrict__ bias,
                                            __bf16* __restrict__ rf) {
  __shared__ __align__(16) __bf16 sW[128 * LDK128];
  STAGE_W(sW, Wt, 128, 16, 17, 128, 5);
  __syncthreads();
  const int lane = threadIdx.x & 31;
  const int wave = threadIdx.x >> 5;
  const int atom0 = (blockIdx.x * 4 + wave) * 16;
  if (atom0 >= N_ATOMS) return;
  const int half = lane >> 4, col0 = lane & 15;
  v8f acc[8];
#pragma unroll
  for (int n = 0; n < 8; ++n) acc[n] = v8f_zero();
#pragma unroll
  for (int kc = 0; kc < 4; ++kc) {
    v16bf a = load_A_f32(r + (size_t)atom0 * 128 + kc * 32, 128, lane);
#pragma unroll
    for (int n = 0; n < 8; ++n) {
      v16bf b = load_B_T(sW, LDK128, n * 16, kc * 32, lane);
      acc[n] = wmma_bf16(a, b, acc[n]);
    }
  }
#pragma unroll
  for (int n = 0; n < 8; ++n) {
    const int col = col0 + n * 16;
    const float bv = bias[col];
#pragma unroll
    for (int i = 0; i < 8; ++i) {
      const int row = i + 8 * half;
      rf[(size_t)(atom0 + row) * 128 + col] = (__bf16)(acc[n][i] + bv);
    }
  }
}

// ---- fused edge kernel: ef = ssp(gauss@We1+b1)@We2+b2; scatter messages ----

__global__ void __launch_bounds__(128) k_edge(const __bf16* __restrict__ gauss,
                                              const __bf16* __restrict__ W1t,
                                              const float* __restrict__ b1,
                                              const __bf16* __restrict__ W2t,
                                              const float* __restrict__ b2,
                                              const __bf16* __restrict__ rf,
                                              const int* __restrict__ nbr,
                                              float* __restrict__ agg) {
  __shared__ __align__(16) __bf16 sW1[128 * LDK32];       // 10.0 KB
  __shared__ __align__(16) __bf16 sW2[128 * LDK128];      // 34.0 KB
  __shared__ __align__(16) __bf16 sT[4][16 * LDK128];     // 17.0 KB
  STAGE_W(sW1, W1t, 128, 4, 5, 32, 3);
  STAGE_W(sW2, W2t, 128, 16, 17, 128, 5);
  __syncthreads();
  const int lane = threadIdx.x & 31;
  const int wave = threadIdx.x >> 5;
  const int e0 = (blockIdx.x * 4 + wave) * 16;  // N_EDGES = 64*9375, exact
  const int half = lane >> 4, col0 = lane & 15;

  // GEMM1: (16x32) @ (32x128)
  v16bf a = load_A_bf16(gauss + (size_t)e0 * N_GAUSS, N_GAUSS, lane);
  v8f acc[8];
#pragma unroll
  for (int n = 0; n < 8; ++n) acc[n] = v8f_zero();
#pragma unroll
  for (int n = 0; n < 8; ++n) {
    v16bf b = load_B_T(sW1, LDK32, n * 16, 0, lane);
    acc[n] = wmma_bf16(a, b, acc[n]);
  }
  __bf16* t = sT[wave];
#pragma unroll
  for (int n = 0; n < 8; ++n) {
    const int col = col0 + n * 16;
    const float bv = b1[col];
#pragma unroll
    for (int i = 0; i < 8; ++i) {
      const int row = i + 8 * half;
      t[row * LDK128 + col] = (__bf16)ssp_f(acc[n][i] + bv);
    }
  }
  __syncthreads();

  // GEMM2: (16x128) @ (128x128)
  v8f ef[8];
#pragma unroll
  for (int n = 0; n < 8; ++n) ef[n] = v8f_zero();
#pragma unroll
  for (int kc = 0; kc < 4; ++kc) {
    v16bf a2 = load_A_bf16(t + kc * 32, LDK128, lane);
#pragma unroll
    for (int n = 0; n < 8; ++n) {
      v16bf b = load_B_T(sW2, LDK128, n * 16, kc * 32, lane);
      ef[n] = wmma_bf16(a2, b, ef[n]);
    }
  }

  // gather rf[a0], rf[a1]; scatter m_ij -> agg[a1], m_ji -> agg[a0]
  int ia0[8], ia1[8];
#pragma unroll
  for (int i = 0; i < 8; ++i) {
    const int e = e0 + i + 8 * half;
    ia0[i] = nbr[2 * e];
    ia1[i] = nbr[2 * e + 1];
  }
#pragma unroll
  for (int n = 0; n < 8; ++n) {
    const int col = col0 + n * 16;
    const float bv = b2[col];
#pragma unroll
    for (int i = 0; i < 8; ++i) {
      const float efv = ef[n][i] + bv;
      const float r0 = (float)rf[(size_t)ia0[i] * 128 + col];
      const float r1 = (float)rf[(size_t)ia1[i] * 128 + col];
      atomicAdd(&agg[(size_t)ia1[i] * 128 + col], r0 * efv);
      atomicAdd(&agg[(size_t)ia0[i] * 128 + col], r1 * efv);
    }
  }
}

// ---- update: r += ssp(agg@Wu1+bu1)@Wu2+bu2 (single weight buffer, Wu1 then Wu2) ----

__global__ void __launch_bounds__(128) k_update(const float* __restrict__ agg,
                                                const __bf16* __restrict__ W1t,
                                                const float* __restrict__ b1,
                                                const __bf16* __restrict__ W2t,
                                                const float* __restrict__ b2,
                                                float* __restrict__ r) {
  __shared__ __align__(16) __bf16 sW[128 * LDK128];   // reused Wu1 then Wu2
  __shared__ __align__(16) __bf16 sT[4][16 * LDK128];
  const int lane = threadIdx.x & 31;
  const int wave = threadIdx.x >> 5;
  const int atom0 = (blockIdx.x * 4 + wave) * 16;
  const bool active = atom0 < N_ATOMS;
  const int half = lane >> 4, col0 = lane & 15;
  STAGE_W(sW, W1t, 128, 16, 17, 128, 5);
  __syncthreads();
  if (active) {
    v8f acc[8];
#pragma unroll
    for (int n = 0; n < 8; ++n) acc[n] = v8f_zero();
#pragma unroll
    for (int kc = 0; kc < 4; ++kc) {
      v16bf a = load_A_f32(agg + (size_t)atom0 * 128 + kc * 32, 128, lane);
#pragma unroll
      for (int n = 0; n < 8; ++n) {
        v16bf b = load_B_T(sW, LDK128, n * 16, kc * 32, lane);
        acc[n] = wmma_bf16(a, b, acc[n]);
      }
    }
    __bf16* t = sT[wave];
#pragma unroll
    for (int n = 0; n < 8; ++n) {
      const int col = col0 + n * 16;
      const float bv = b1[col];
#pragma unroll
      for (int i = 0; i < 8; ++i) {
        const int row = i + 8 * half;
        t[row * LDK128 + col] = (__bf16)ssp_f(acc[n][i] + bv);
      }
    }
  }
  __syncthreads();
  STAGE_W(sW, W2t, 128, 16, 17, 128, 5);
  __syncthreads();
  if (active) {
    v8f acc[8];
#pragma unroll
    for (int n = 0; n < 8; ++n) acc[n] = v8f_zero();
#pragma unroll
    for (int kc = 0; kc < 4; ++kc) {
      v16bf a2 = load_A_bf16(sT[wave] + kc * 32, LDK128, lane);
#pragma unroll
      for (int n = 0; n < 8; ++n) {
        v16bf b = load_B_T(sW, LDK128, n * 16, kc * 32, lane);
        acc[n] = wmma_bf16(a2, b, acc[n]);
      }
    }
#pragma unroll
    for (int n = 0; n < 8; ++n) {
      const int col = col0 + n * 16;
      const float bv = b2[col];
#pragma unroll
      for (int i = 0; i < 8; ++i) {
        const int row = i + 8 * half;
        r[(size_t)(atom0 + row) * 128 + col] += acc[n][i] + bv;
      }
    }
  }
}

// ---- readout: e_mol[mol][k] += ssp(r@Wr1[k]+br1)@Wr2[k]+br2 ----

__global__ void __launch_bounds__(128) k_readout(const float* __restrict__ r,
                                                 const __bf16* __restrict__ W1t,
                                                 const float* __restrict__ br1,
                                                 const float* __restrict__ W2,
                                                 const float* __restrict__ br2,
                                                 const int* __restrict__ mol_idx,
                                                 float* __restrict__ out) {
  __shared__ __align__(16) __bf16 sW[2 * 64 * LDK128];  // both keys, 34 KB
  __shared__ float sRed[4 * 16];
  const int lane = threadIdx.x & 31;
  const int wave = threadIdx.x >> 5;
  const int atom0 = (blockIdx.x * 4 + wave) * 16;
  const bool active = atom0 < N_ATOMS;
  const int half = lane >> 4, col0 = lane & 15;
  STAGE_W(sW, W1t, 128, 16, 17, 128, 5);  // 2*64 rows of K=128
  for (int k = 0; k < 2; ++k) {
    if (threadIdx.x < 64) sRed[threadIdx.x] = 0.0f;
    __syncthreads();
    if (active) {
      v8f acc[4];
#pragma unroll
      for (int n = 0; n < 4; ++n) acc[n] = v8f_zero();
#pragma unroll
      for (int kc = 0; kc < 4; ++kc) {
        v16bf a = load_A_f32(r + (size_t)atom0 * 128 + kc * 32, 128, lane);
#pragma unroll
        for (int n = 0; n < 4; ++n) {
          v16bf b = load_B_T(sW + k * 64 * LDK128, LDK128, n * 16, kc * 32, lane);
          acc[n] = wmma_bf16(a, b, acc[n]);
        }
      }
      float p[8] = {0.f, 0.f, 0.f, 0.f, 0.f, 0.f, 0.f, 0.f};
#pragma unroll
      for (int n = 0; n < 4; ++n) {
        const int col = col0 + n * 16;
        const float bv = br1[k * 64 + col];
        const float wv = W2[k * 64 + col];
#pragma unroll
        for (int i = 0; i < 8; ++i) p[i] += ssp_f(acc[n][i] + bv) * wv;
      }
#pragma unroll
      for (int i = 0; i < 8; ++i)
        atomicAdd(&sRed[wave * 16 + i + 8 * half], p[i]);
    }
    __syncthreads();
    if (active && lane < 16) {
      const int atom = atom0 + lane;
      const float e = sRed[wave * 16 + lane] + br2[k];
      atomicAdd(&out[mol_idx[atom] * 2 + k], e);
    }
    __syncthreads();
  }
}

// ---------------- host launcher ----------------

extern "C" void kernel_launch(void* const* d_in, const int* in_sizes, int n_in,
                              void* d_out, int out_size, void* d_ws, size_t ws_size,
                              hipStream_t stream) {
  (void)in_sizes; (void)n_in; (void)out_size; (void)ws_size;
  const int*   z     = (const int*)d_in[0];
  const float* xyz   = (const float*)d_in[1];
  const int*   nbr   = (const int*)d_in[2];
  const int*   molix = (const int*)d_in[3];
  const float* embed = (const float*)d_in[4];
  const float* We1   = (const float*)d_in[5];
  const float* be1   = (const float*)d_in[6];
  const float* We2   = (const float*)d_in[7];
  const float* be2   = (const float*)d_in[8];
  const float* Wn    = (const float*)d_in[9];
  const float* bn    = (const float*)d_in[10];
  const float* Wu1   = (const float*)d_in[11];
  const float* bu1   = (const float*)d_in[12];
  const float* Wu2   = (const float*)d_in[13];
  const float* bu2   = (const float*)d_in[14];
  const float* Wr1   = (const float*)d_in[15];
  const float* br1   = (const float*)d_in[16];
  const float* Wr2   = (const float*)d_in[17];
  const float* br2   = (const float*)d_in[18];
  float* out = (float*)d_out;

  uint8_t* ws = (uint8_t*)d_ws;
  size_t off = 0;
  auto carve = [&](size_t bytes) -> void* {
    void* p = ws + off;
    off = (off + bytes + 255) & ~(size_t)255;
    return p;
  };
  float*  r_buf = (float*)carve((size_t)N_ATOMS * 128 * 4);
  float*  agg   = (float*)carve((size_t)N_ATOMS * 128 * 4);
  __bf16* rf    = (__bf16*)carve((size_t)N_ATOMS * 128 * 2);
  __bf16* gauss = (__bf16*)carve((size_t)N_EDGES * N_GAUSS * 2);
  __bf16* wWe1  = (__bf16*)carve((size_t)3 * 32 * 128 * 2);   // (3,128,32) T
  __bf16* wWe2  = (__bf16*)carve((size_t)3 * 128 * 128 * 2);  // (3,128,128) T
  __bf16* wWn   = (__bf16*)carve((size_t)3 * 128 * 128 * 2);
  __bf16* wWu1  = (__bf16*)carve((size_t)3 * 128 * 128 * 2);
  __bf16* wWu2  = (__bf16*)carve((size_t)3 * 128 * 128 * 2);
  __bf16* wWr1  = (__bf16*)carve((size_t)2 * 128 * 64 * 2);   // (2,64,128) T

  auto cvt_t = [&](const float* s, __bf16* d, int K, int N, int L) {
    int total = L * K * N;
    k_cvt_t<<<(total + 255) / 256, 256, 0, stream>>>(s, d, K, N, total);
  };
  cvt_t(We1, wWe1, 32, 128, 3);
  cvt_t(We2, wWe2, 128, 128, 3);
  cvt_t(Wn,  wWn,  128, 128, 3);
  cvt_t(Wu1, wWu1, 128, 128, 3);
  cvt_t(Wu2, wWu2, 128, 128, 3);
  cvt_t(Wr1, wWr1, 128, 64, 2);

  k_init_r<<<(N_ATOMS * 128 + 255) / 256, 256, 0, stream>>>(z, embed, r_buf);
  k_gauss<<<(N_EDGES + 255) / 256, 256, 0, stream>>>(xyz, nbr, gauss);

  const int ATOM_BLOCKS = (N_ATOMS / 16 + 3) / 4;  // 782
  const int EDGE_BLOCKS = N_EDGES / 64;            // 9375
  for (int l = 0; l < 3; ++l) {
    k_rf<<<ATOM_BLOCKS, 128, 0, stream>>>(r_buf, wWn + (size_t)l * 128 * 128,
                                          bn + l * 128, rf);
    k_zero<<<(N_ATOMS * 128 + 255) / 256, 256, 0, stream>>>(agg, N_ATOMS * 128);
    k_edge<<<EDGE_BLOCKS, 128, 0, stream>>>(
        gauss, wWe1 + (size_t)l * 128 * 32, be1 + l * 128,
        wWe2 + (size_t)l * 128 * 128, be2 + l * 128, rf, nbr, agg);
    k_update<<<ATOM_BLOCKS, 128, 0, stream>>>(
        agg, wWu1 + (size_t)l * 128 * 128, bu1 + l * 128,
        wWu2 + (size_t)l * 128 * 128, bu2 + l * 128, r_buf);
  }
  k_zero<<<(N_MOL * 2 + 255) / 256, 256, 0, stream>>>(out, N_MOL * 2);
  k_readout<<<ATOM_BLOCKS, 128, 0, stream>>>(r_buf, wWr1, br1, Wr2, br2, molix, out);
}
